// GraphEncoder_19971597926820
// MI455X (gfx1250) — matrix operand order repaired and checked by reading, not compiled, not run on hardware
//
#include <hip/hip_runtime.h>
#include <hip/hip_bf16.h>

typedef __attribute__((ext_vector_type(16))) __bf16 v16bf;
typedef __attribute__((ext_vector_type(8)))  float  v8f;

__device__ __forceinline__ float lrelu(float x) { return x > 0.0f ? x : 0.2f * x; }

__device__ __forceinline__ void atomicMaxF(float* addr, float val) {
    unsigned int* ua = reinterpret_cast<unsigned int*>(addr);
    unsigned int cur = *ua;
    while (__uint_as_float(cur) < val) {
        unsigned int prev = atomicCAS(ua, cur, __float_as_uint(val));
        if (prev == cur) break;
        cur = prev;
    }
}

// ---------------------------------------------------------------------------
// WMMA GEMM specialized for K == 128 (all GEMMs in this model):
// Y[nrows x ncols] = X[nrows x 128] @ W[128 x ncols] (+bias)(+relu).
// One wave per 16-row strip x 64-column group. The wave loads the full
// 16x128 A strip ONCE into 4 bf16 fragments (32 VGPRs), then sweeps 4
// column tiles -> 16 static v_wmma_f32_16x16x32_bf16 per wave, A traffic
// reduced to ncols/64 redundancy.
// ---------------------------------------------------------------------------
__global__ __launch_bounds__(32)
void wmma_gemm_k128(const float* __restrict__ X, const float* __restrict__ W,
                    const float* __restrict__ bias, float* __restrict__ Y,
                    int nrows, int ncols, int do_relu) {
    constexpr int K = 128;
    const int lane = threadIdx.x & 31;
    const int m    = lane & 15;
    const int grp  = lane >> 4;              // 0 or 1
    const int rowTile = blockIdx.x * 16;
    const int colBase = blockIdx.y * 64;     // 4 tiles of 16 columns
    const int arow = rowTile + m;

    // ---- load whole A strip (16 x 128) as 4 fragments ----
    v16bf afrag[4];
    #pragma unroll
    for (int kk = 0; kk < 4; ++kk) {
        v16bf a = {};
        if (arow < nrows) {
            const float* xr = X + (size_t)arow * K + kk * 32 + grp * 8;
            #pragma unroll
            for (int v = 0; v < 4; ++v) {
                a[2 * v]     = (__bf16)xr[2 * v];
                a[2 * v + 1] = (__bf16)xr[2 * v + 1];
                a[8 + 2 * v] = (__bf16)xr[16 + 2 * v];
                a[9 + 2 * v] = (__bf16)xr[16 + 2 * v + 1];
            }
        }
        afrag[kk] = a;
    }

    // ---- sweep 4 column tiles ----
    #pragma unroll
    for (int t = 0; t < 4; ++t) {
        const int bcol = colBase + t * 16 + m;   // B column held by this lane
        v8f acc = {};
        #pragma unroll
        for (int kk = 0; kk < 4; ++kk) {
            v16bf b = {};
            // B fragment (32x16): lanes 0-15 hold K=k0..k0+7,k0+16..k0+23, lanes 16-31 K=k0+8..k0+15,k0+24..k0+31
            const float* wc = W + (size_t)(kk * 32 + grp * 8) * ncols + bcol;
            #pragma unroll
            for (int v = 0; v < 4; ++v) {
                b[2 * v]     = (__bf16)wc[(size_t)(2 * v) * ncols];
                b[2 * v + 1] = (__bf16)wc[(size_t)(2 * v + 1) * ncols];
                b[8 + 2 * v] = (__bf16)wc[(size_t)(16 + 2 * v) * ncols];
                b[9 + 2 * v] = (__bf16)wc[(size_t)(16 + 2 * v + 1) * ncols];
            }
            acc = __builtin_amdgcn_wmma_f32_16x16x32_bf16(
                false, afrag[kk], false, b, (short)0, acc, false, false);
        }
        const float bv = bias ? bias[bcol] : 0.0f;
        // D layout: VGPR v -> row (rowTile + v + 8*grp), col bcol
        #pragma unroll
        for (int v = 0; v < 8; ++v) {
            int r = rowTile + v + 8 * grp;
            if (r < nrows) {
                float val = acc[v] + bv;
                if (do_relu) val = fmaxf(val, 0.0f);
                Y[(size_t)r * ncols + bcol] = val;
            }
        }
    }
}

// ---------------------------------------------------------------------------
// Elementwise / graph kernels
// ---------------------------------------------------------------------------
__global__ void k_fill(float* p, float v, int n) {
    int i = blockIdx.x * blockDim.x + threadIdx.x;
    if (i < n) p[i] = v;
}

__global__ void k_deg_edges(const int* __restrict__ dst, float* __restrict__ deg, int E) {
    int e = blockIdx.x * blockDim.x + threadIdx.x;
    if (e < E) atomicAdd(&deg[dst[e]], 1.0f);
}

__global__ void k_rsqrt(float* p, int n) {
    int i = blockIdx.x * blockDim.x + threadIdx.x;
    if (i < n) p[i] = rsqrtf(p[i]);
}

// out[i][c] = h[i][c] * dinv[i]^2   (self-loop term of GCN aggregation)
__global__ void k_gcn_self(const float* __restrict__ h, const float* __restrict__ dinv,
                           float* __restrict__ out, int n, int C) {
    int idx = blockIdx.x * blockDim.x + threadIdx.x;
    if (idx < n * C) {
        int i = idx / C;
        float d = dinv[i];
        out[idx] = h[idx] * d * d;
    }
}

// one thread per (edge, 4-feature chunk)
__global__ void k_gcn_edges(const int* __restrict__ src, const int* __restrict__ dst,
                            const float* __restrict__ h, const float* __restrict__ dinv,
                            float* __restrict__ out, int E, int C) {
    long long tid = (long long)blockIdx.x * blockDim.x + threadIdx.x;
    int chunks = C / 4;
    if (tid >= (long long)E * chunks) return;
    int e  = (int)(tid / chunks);
    int f4 = (int)(tid % chunks) * 4;
    int s = src[e], d = dst[e];
    float nrm = dinv[s] * dinv[d];
    const float4 hs = *(const float4*)(h + (size_t)s * C + f4);
    float* od = out + (size_t)d * C + f4;
    atomicAdd(&od[0], hs.x * nrm);
    atomicAdd(&od[1], hs.y * nrm);
    atomicAdd(&od[2], hs.z * nrm);
    atomicAdd(&od[3], hs.w * nrm);
}

__global__ void k_bias_relu(float* __restrict__ p, const float* __restrict__ b, int n, int C) {
    int idx = blockIdx.x * blockDim.x + threadIdx.x;
    if (idx < n * C) p[idx] = fmaxf(p[idx] + b[idx % C], 0.0f);
}

// per (node, head): a_src/a_dst dot products over 128 channels
__global__ void k_gat_att(const float* __restrict__ t, const float* __restrict__ ws,
                          const float* __restrict__ wd, float* __restrict__ as,
                          float* __restrict__ ad, int n, int H, int C) {
    int idx = blockIdx.x * blockDim.x + threadIdx.x;
    if (idx >= n * H) return;
    int i = idx / H, h = idx % H;
    const float* ti = t + (size_t)i * H * C + (size_t)h * C;
    const float* s = ws + (size_t)h * C;
    const float* d = wd + (size_t)h * C;
    float vs = 0.0f, vd = 0.0f;
    for (int c = 0; c < C; ++c) { vs += ti[c] * s[c]; vd += ti[c] * d[c]; }
    as[idx] = vs; ad[idx] = vd;
}

// emax[i][h] = lrelu(a_src[i][h] + a_dst[i][h])  (self-loop seeds the max)
__global__ void k_emax_init(const float* __restrict__ as, const float* __restrict__ ad,
                            float* __restrict__ emax, int nH) {
    int idx = blockIdx.x * blockDim.x + threadIdx.x;
    if (idx < nH) emax[idx] = lrelu(as[idx] + ad[idx]);
}

__global__ void k_emax_edges(const int* __restrict__ src, const int* __restrict__ dst,
                             const float* __restrict__ as, const float* __restrict__ ad,
                             float* __restrict__ emax, int E, int H) {
    int tid = blockIdx.x * blockDim.x + threadIdx.x;
    if (tid >= E * H) return;
    int e = tid / H, h = tid % H;
    int s = src[e], d = dst[e];
    atomicMaxF(&emax[d * H + h], lrelu(as[s * H + h] + ad[d * H + h]));
}

__global__ void k_denom_init(const float* __restrict__ as, const float* __restrict__ ad,
                             const float* __restrict__ emax, float* __restrict__ den, int nH) {
    int idx = blockIdx.x * blockDim.x + threadIdx.x;
    if (idx < nH) den[idx] = expf(lrelu(as[idx] + ad[idx]) - emax[idx]);
}

__global__ void k_denom_edges(const int* __restrict__ src, const int* __restrict__ dst,
                              const float* __restrict__ as, const float* __restrict__ ad,
                              const float* __restrict__ emax, float* __restrict__ den,
                              int E, int H) {
    int tid = blockIdx.x * blockDim.x + threadIdx.x;
    if (tid >= E * H) return;
    int e = tid / H, h = tid % H;
    int s = src[e], d = dst[e];
    float ev = lrelu(as[s * H + h] + ad[d * H + h]);
    atomicAdd(&den[d * H + h], expf(ev - emax[d * H + h]));
}

// materialize per-edge softmax weights once: alpha[e][h]
__global__ void k_alpha_edges(const int* __restrict__ src, const int* __restrict__ dst,
                              const float* __restrict__ as, const float* __restrict__ ad,
                              const float* __restrict__ emax, const float* __restrict__ den,
                              float* __restrict__ alpha, int E, int H) {
    int tid = blockIdx.x * blockDim.x + threadIdx.x;
    if (tid >= E * H) return;
    int e = tid / H, h = tid % H;
    int s = src[e], d = dst[e];
    float ev = lrelu(as[s * H + h] + ad[d * H + h]);
    alpha[tid] = expf(ev - emax[d * H + h]) / den[d * H + h];
}

// acc[i][c] = (1/H) * sum_h alpha_self(i,h) * t[i][h][c]
__global__ void k_agg_init(const float* __restrict__ t, const float* __restrict__ as,
                           const float* __restrict__ ad, const float* __restrict__ emax,
                           const float* __restrict__ den, float* __restrict__ acc,
                           int n, int H, int C) {
    int idx = blockIdx.x * blockDim.x + threadIdx.x;
    if (idx >= n * C) return;
    int i = idx / C, c = idx % C;
    float v = 0.0f;
    for (int h = 0; h < H; ++h) {
        float ev = lrelu(as[i * H + h] + ad[i * H + h]);
        float a  = expf(ev - emax[i * H + h]) / den[i * H + h];
        v += a * t[(size_t)i * H * C + (size_t)h * C + c];
    }
    acc[idx] = v / (float)H;
}

// one thread per (edge, 4-feature chunk): acc[d][c] += (1/H) sum_h alpha[e][h] * t[s][h][c]
__global__ void k_agg_edges(const int* __restrict__ src, const int* __restrict__ dst,
                            const float* __restrict__ t, const float* __restrict__ alpha,
                            float* __restrict__ acc, int E, int H, int C) {
    long long tid = (long long)blockIdx.x * blockDim.x + threadIdx.x;
    int chunks = C / 4;
    if (tid >= (long long)E * chunks) return;
    int e  = (int)(tid / chunks);
    int f4 = (int)(tid % chunks) * 4;
    int s = src[e], d = dst[e];
    const float4 al = *(const float4*)(alpha + (size_t)e * 4);
    const float* ts = t + (size_t)s * H * C + f4;
    float* od = acc + (size_t)d * C + f4;
    const float4 t0 = *(const float4*)(ts + 0 * C);
    const float4 t1 = *(const float4*)(ts + 1 * C);
    const float4 t2 = *(const float4*)(ts + 2 * C);
    const float4 t3 = *(const float4*)(ts + 3 * C);
    atomicAdd(&od[0], 0.25f * (al.x * t0.x + al.y * t1.x + al.z * t2.x + al.w * t3.x));
    atomicAdd(&od[1], 0.25f * (al.x * t0.y + al.y * t1.y + al.z * t2.y + al.w * t3.y));
    atomicAdd(&od[2], 0.25f * (al.x * t0.z + al.y * t1.z + al.z * t2.z + al.w * t3.z));
    atomicAdd(&od[3], 0.25f * (al.x * t0.w + al.y * t1.w + al.z * t2.w + al.w * t3.w));
}

// ---------------------------------------------------------------------------
extern "C" void kernel_launch(void* const* d_in, const int* in_sizes, int n_in,
                              void* d_out, int out_size, void* d_ws, size_t ws_size,
                              hipStream_t stream) {
    const float* x       = (const float*)d_in[0];
    const int*   edge    = (const int*)d_in[1];
    const float* gcn_w   = (const float*)d_in[2];
    const float* gcn_b   = (const float*)d_in[3];
    const float* gat_w   = (const float*)d_in[4];
    const float* att_src = (const float*)d_in[5];
    const float* att_dst = (const float*)d_in[6];
    const float* gat_b   = (const float*)d_in[7];
    const float* mlp_w1  = (const float*)d_in[8];
    const float* mlp_b1  = (const float*)d_in[9];
    const float* mlp_w2  = (const float*)d_in[10];
    const float* mlp_b2  = (const float*)d_in[11];
    const float* mu_w    = (const float*)d_in[12];
    const float* mu_b    = (const float*)d_in[13];
    const float* lv_w    = (const float*)d_in[14];
    const float* lv_b    = (const float*)d_in[15];

    const int N = 10000, HID = 128, Z = 64, H = 4;
    const int E = in_sizes[1] / 2;
    const int* srcIdx = edge;
    const int* dstIdx = edge + E;

    float* ws    = (float*)d_ws;
    float* tmpA  = ws;                             // N*512 : GAT transformed feats
    float* bufB  = tmpA + (size_t)N * 512;         // N*128
    float* bufC  = bufB + (size_t)N * 128;         // N*128
    float* bufD  = bufC + (size_t)N * 128;         // N*128
    float* dinv  = bufD + (size_t)N * 128;         // N
    float* asrc  = dinv + N;                       // N*H
    float* adst  = asrc + (size_t)N * H;           // N*H
    float* emax  = adst + (size_t)N * H;           // N*H
    float* denm  = emax + (size_t)N * H;           // N*H
    float* alpha = denm + (size_t)N * H;           // E*H

    const int rowT = (N + 15) / 16;
    auto nb = [](long long t) { return dim3((unsigned)((t + 255) / 256)); };
    dim3 b256(256), b32(32);

    // ---- GCN ----
    // h0 = x @ gcn_w
    wmma_gemm_k128<<<dim3(rowT, HID / 64), b32, 0, stream>>>(x, gcn_w, nullptr, bufB, N, HID, 0);
    // degrees (self-loop => init 1.0), then dinv = rsqrt(deg)
    k_fill<<<nb(N), b256, 0, stream>>>(dinv, 1.0f, N);
    k_deg_edges<<<nb(E), b256, 0, stream>>>(dstIdx, dinv, E);
    k_rsqrt<<<nb(N), b256, 0, stream>>>(dinv, N);
    // aggregation: self term, edge scatter, bias+relu
    k_gcn_self<<<nb((long long)N * HID), b256, 0, stream>>>(bufB, dinv, bufC, N, HID);
    k_gcn_edges<<<nb((long long)E * (HID / 4)), b256, 0, stream>>>(srcIdx, dstIdx, bufB, dinv, bufC, E, HID);
    k_bias_relu<<<nb((long long)N * HID), b256, 0, stream>>>(bufC, gcn_b, N, HID);

    // ---- GAT ----
    // t = h @ gat_w  (N x 512)
    wmma_gemm_k128<<<dim3(rowT, (H * HID) / 64), b32, 0, stream>>>(bufC, gat_w, nullptr, tmpA, N, H * HID, 0);
    k_gat_att<<<nb((long long)N * H), b256, 0, stream>>>(tmpA, att_src, att_dst, asrc, adst, N, H, HID);
    // segment softmax (self-loop seeds both max and denom)
    k_emax_init<<<nb((long long)N * H), b256, 0, stream>>>(asrc, adst, emax, N * H);
    k_emax_edges<<<nb((long long)E * H), b256, 0, stream>>>(srcIdx, dstIdx, asrc, adst, emax, E, H);
    k_denom_init<<<nb((long long)N * H), b256, 0, stream>>>(asrc, adst, emax, denm, N * H);
    k_denom_edges<<<nb((long long)E * H), b256, 0, stream>>>(srcIdx, dstIdx, asrc, adst, emax, denm, E, H);
    k_alpha_edges<<<nb((long long)E * H), b256, 0, stream>>>(srcIdx, dstIdx, asrc, adst, emax, denm, alpha, E, H);
    // weighted aggregation, head-mean folded in; then bias+relu
    k_agg_init<<<nb((long long)N * HID), b256, 0, stream>>>(tmpA, asrc, adst, emax, denm, bufD, N, H, HID);
    k_agg_edges<<<nb((long long)E * (HID / 4)), b256, 0, stream>>>(srcIdx, dstIdx, tmpA, alpha, bufD, E, H, HID);
    k_bias_relu<<<nb((long long)N * HID), b256, 0, stream>>>(bufD, gat_b, N, HID);

    // ---- MLP ----
    wmma_gemm_k128<<<dim3(rowT, HID / 64), b32, 0, stream>>>(bufD, mlp_w1, mlp_b1, bufB, N, HID, 1);
    wmma_gemm_k128<<<dim3(rowT, HID / 64), b32, 0, stream>>>(bufB, mlp_w2, mlp_b2, bufC, N, HID, 1);

    // ---- heads: mu then logvar concatenated in d_out ----
    float* outp = (float*)d_out;
    wmma_gemm_k128<<<dim3(rowT, Z / 64), b32, 0, stream>>>(bufC, mu_w, mu_b, outp, N, Z, 0);
    wmma_gemm_k128<<<dim3(rowT, Z / 64), b32, 0, stream>>>(bufC, lv_w, lv_b, outp + (size_t)N * Z, N, Z, 0);
}
